// GAT_48129403519483
// MI455X (gfx1250) — matrix operand order repaired and checked by reading, not compiled, not run on hardware
//
#include <hip/hip_runtime.h>
#include <hip/hip_bf16.h>
#include <math.h>

// ---------------------------------------------------------------------------
// GAT (2 layers, 2 slots) for MI455X / gfx1250, wave32 + WMMA bf16.
// B=4, N=2048, F=128.
// Roofline: outputs (4 x [B,N,N] attentions + 2 embeddings) = ~268 MB written;
// the attention GEMM re-reads them (mostly from 192MB L2). Compute (17 GFLOP
// bf16) is negligible vs the WMMA ceiling -> memory-bound. Strategy:
//  * attention read exactly once per GEMM (16x128 wave tile, 8 WMMA / A-frag)
//  * B operands pre-transposed + pre-converted to bf16 so each lane's WMMA
//    B-fragment is a 32B contiguous load (2x global_load_b128, immediate offs)
//  * k-loop ping-pong double buffering; __launch_bounds__(256,2) gives the
//    allocator room (~230 VGPRs live) so both buffers stay in registers and
//    loads overlap WMMA issue (deep per-wave pipeline > thin high occupancy)
// ---------------------------------------------------------------------------

typedef __attribute__((ext_vector_type(16))) __bf16 v16bf;
typedef __attribute__((ext_vector_type(8)))  float  v8f;

#define GAT_B 4
#define GAT_N 2048
#define GAT_F 128
#define GAT_ALPHA 0.2f

static __device__ __forceinline__ __bf16 f2bf(float x) { return (__bf16)x; }

static __device__ __forceinline__ v16bf load_a_frag(const float* __restrict__ ap)
{
    v16bf af;
#pragma unroll
    for (int i = 0; i < 8; ++i) af[i] = f2bf(ap[i]);
#pragma unroll
    for (int i = 0; i < 8; ++i) af[8 + i] = f2bf(ap[8 + i]);
    return af;
}

// ---------------------------------------------------------------------------
// Batched GEMM:  C[b] = act( A[b] (MxK f32, row-major) @ B[b] (KxNc) )
// B is supplied TRANSPOSED in bf16: BT[b] is [Nc][K], K-contiguous.
// One wave computes a 16x128 output tile (Nc == 128): one A fragment feeds
// 8 WMMAs; each B fragment is one contiguous 32B load per lane at an
// immediate offset (K is compile-time).  K % 64 == 0.
// Fragment layouts used here (wave32; A/B in this code's contiguous k order):
//   A 16x32 bf16 : lane<16 holds M=lane, elem i -> K=k0+i ;
//                  lanes 16..31 same M, K=k0+16+i (matches B).
//   B 32x16 bf16 : lane<16 -> N=lane, elem i -> K=k0+i ; lanes 16..31 K=k0+16+i
//                  == BT[n][k0 + 16*hi + i]  (contiguous in i).
//   C/D 16x16 f32: VGPR r, lane<16 -> M=r, N=lane ; lane>=16 -> M=8+r.
// ---------------------------------------------------------------------------
template <int K, bool RELU>
__global__ __launch_bounds__(256, 2)
void wmma_gemm_bf16T(const float* __restrict__ A,
                     const __bf16* __restrict__ BT,
                     float* __restrict__ C,
                     int M, int Nc,
                     long sAb, long sBTb, long sCb)
{
    const int lane = threadIdx.x & 31;
    const int wave = blockIdx.x * (blockDim.x >> 5) + (threadIdx.x >> 5);

    const int perBatch = M >> 4;                 // 16-row tiles per batch
    const int b     = wave / perBatch;
    const int mTile = wave - b * perBatch;

    const float*  Ab = A  + (long)b * sAb;
    const __bf16* Bb = BT + (long)b * sBTb;
    float*        Cb = C  + (long)b * sCb;

    const int row0 = mTile << 4;
    const int hi   = lane >> 4;    // 0: lanes 0-15, 1: lanes 16-31
    const int lid  = lane & 15;

    const float*  arow = Ab + (long)(row0 + lid) * K + hi * 16;
    const __bf16* bbase = Bb + (long)lid * K + hi * 16;   // + t*16*K + k0 (imm)

    v8f acc[8];
#pragma unroll
    for (int t = 0; t < 8; ++t) acc[t] = (v8f){};

    // ---- prologue: fragments for k0 = 0 ----
    v16bf a0 = load_a_frag(arow);
    v16bf b0[8];
#pragma unroll
    for (int t = 0; t < 8; ++t)
        b0[t] = *(const v16bf*)(bbase + (long)t * 16 * K);

    v16bf a1;
    v16bf b1[8];

    for (int k0 = 0; k0 < K; k0 += 64) {
        // load k0+32 fragments (ping buffer 1) while WMMAs on buffer 0 issue
        a1 = load_a_frag(arow + k0 + 32);
#pragma unroll
        for (int t = 0; t < 8; ++t)
            b1[t] = *(const v16bf*)(bbase + (long)t * 16 * K + k0 + 32);

#pragma unroll
        for (int t = 0; t < 8; ++t)
            acc[t] = __builtin_amdgcn_wmma_f32_16x16x32_bf16(
                false, a0, false, b0[t], (short)0, acc[t], false, false);

        if (k0 + 64 < K) {
            a0 = load_a_frag(arow + k0 + 64);
#pragma unroll
            for (int t = 0; t < 8; ++t)
                b0[t] = *(const v16bf*)(bbase + (long)t * 16 * K + k0 + 64);
        }

#pragma unroll
        for (int t = 0; t < 8; ++t)
            acc[t] = __builtin_amdgcn_wmma_f32_16x16x32_bf16(
                false, a1, false, b1[t], (short)0, acc[t], false, false);
    }

    // ---- store (C/D layout: VGPR r -> M = r + 8*hi, N = t*16 + lid) ----
#pragma unroll
    for (int r = 0; r < 8; ++r) {
        float* crow = Cb + (long)(row0 + hi * 8 + r) * Nc + lid;
#pragma unroll
        for (int t = 0; t < 8; ++t) {
            float v = acc[t][r];
            if (RELU) v = fmaxf(v, 0.f);
            crow[t * 16] = v;
        }
    }
}

// ---------------------------------------------------------------------------
// Convert + transpose to bf16:  dst[y][c][r] = bf16(src[y][r][c])
// (y = batch via gridDim.y). Reads coalesced; writes small (<= 2MB total).
// ---------------------------------------------------------------------------
__global__ void cvt_transpose_bf16(const float* __restrict__ src,
                                   __bf16* __restrict__ dst,
                                   int R, int Ccols)
{
    const long base = (long)blockIdx.y * R * Ccols;
    const int  idx  = blockIdx.x * blockDim.x + threadIdx.x;   // r*Ccols + c
    const int  r    = idx / Ccols;
    const int  c    = idx - r * Ccols;
    dst[base + (long)c * R + r] = f2bf(src[base + idx]);
}

// ---------------------------------------------------------------------------
// scores: s1[row] = ne_row . a[0:F] + xw_row . a[F:2F]
//         s2[row] = ne_row . a[2F:3F] + xw_row . a[3F:4F]
// One wave32 per row (row = b*N + n).
// ---------------------------------------------------------------------------
__global__ void scores_kernel(const float* __restrict__ ne,
                              const float* __restrict__ xw,
                              const float* __restrict__ a,
                              float* __restrict__ s1,
                              float* __restrict__ s2)
{
    const int F = GAT_F;
    const int row  = blockIdx.x * (blockDim.x >> 5) + (threadIdx.x >> 5);
    const int lane = threadIdx.x & 31;

    const float* nrow = ne + (long)row * F;
    const float* xrow = xw + (long)row * F;

    float acc1 = 0.f, acc2 = 0.f;
#pragma unroll
    for (int i = lane; i < F; i += 32) {
        float nv = nrow[i];
        float xv = xrow[i];
        acc1 += nv * a[i]         + xv * a[F + i];
        acc2 += nv * a[2 * F + i] + xv * a[3 * F + i];
    }
#pragma unroll
    for (int o = 16; o > 0; o >>= 1) {
        acc1 += __shfl_xor(acc1, o, 32);
        acc2 += __shfl_xor(acc2, o, 32);
    }
    if (lane == 0) { s1[row] = acc1; s2[row] = acc2; }
}

// ---------------------------------------------------------------------------
// Block reductions (256 threads = 8 waves)
// ---------------------------------------------------------------------------
static __device__ __forceinline__ float blockMax(float v, float* red)
{
    const int lane = threadIdx.x & 31, wid = threadIdx.x >> 5;
#pragma unroll
    for (int o = 16; o > 0; o >>= 1) v = fmaxf(v, __shfl_xor(v, o, 32));
    if (lane == 0) red[wid] = v;
    __syncthreads();
    float r = red[0];
#pragma unroll
    for (int i = 1; i < 8; ++i) r = fmaxf(r, red[i]);
    __syncthreads();
    return r;
}

static __device__ __forceinline__ float blockSum(float v, float* red)
{
    const int lane = threadIdx.x & 31, wid = threadIdx.x >> 5;
#pragma unroll
    for (int o = 16; o > 0; o >>= 1) v += __shfl_xor(v, o, 32);
    if (lane == 0) red[wid] = v;
    __syncthreads();
    float r = red[0];
#pragma unroll
    for (int i = 1; i < 8; ++i) r += red[i];
    __syncthreads();
    return r;
}

// ---------------------------------------------------------------------------
// attention row: att[b,i,j] = softmax_j( leakyrelu(s1[b,i] + s2[b,j]) )
// One 256-thread block per (b,i) row; 8 elements/thread in registers.
// ---------------------------------------------------------------------------
__global__ void attn_softmax_kernel(const float* __restrict__ s1,
                                    const float* __restrict__ s2,
                                    float* __restrict__ att)
{
    const int N = GAT_N;
    __shared__ float red[8];

    const int row = blockIdx.x;          // b*N + i
    const int b   = row >> 11;           // / 2048
    const int tid = threadIdx.x;

    const float si  = s1[row];
    const float* sb = s2 + (long)b * N;

    float e[8];
    float mx = -1e30f;
#pragma unroll
    for (int k = 0; k < 8; ++k) {
        float v = si + sb[tid + k * 256];
        v = (v >= 0.f) ? v : GAT_ALPHA * v;
        e[k] = v;
        mx = fmaxf(mx, v);
    }
    mx = blockMax(mx, red);

    float sum = 0.f;
#pragma unroll
    for (int k = 0; k < 8; ++k) {
        e[k] = __expf(e[k] - mx);
        sum += e[k];
    }
    sum = blockSum(sum, red);
    const float inv = 1.0f / sum;

    float* out = att + (long)row * N;
#pragma unroll
    for (int k = 0; k < 8; ++k) out[tid + k * 256] = e[k] * inv;
}

// ---------------------------------------------------------------------------
// log_softmax over axis=1 (node dim): one block per (b, f) column.
// ---------------------------------------------------------------------------
__global__ void log_softmax_axis1_kernel(const float* __restrict__ h,
                                         float* __restrict__ out)
{
    const int N = GAT_N, F = GAT_F;
    __shared__ float red[8];

    const int col = blockIdx.x;          // b*F + f
    const int b   = col >> 7;
    const int f   = col & (F - 1);
    const int tid = threadIdx.x;

    const float* base = h + (long)b * N * F + f;

    float v[8];
    float mx = -1e30f;
#pragma unroll
    for (int k = 0; k < 8; ++k) {
        v[k] = base[(long)(tid + k * 256) * F];
        mx = fmaxf(mx, v[k]);
    }
    mx = blockMax(mx, red);

    float sum = 0.f;
#pragma unroll
    for (int k = 0; k < 8; ++k) sum += __expf(v[k] - mx);
    sum = blockSum(sum, red);
    const float lse = mx + __logf(sum);

    float* obase = out + (long)b * N * F + f;
#pragma unroll
    for (int k = 0; k < 8; ++k)
        obase[(long)(tid + k * 256) * F] = v[k] - lse;
}

// ---------------------------------------------------------------------------
// Orchestration
// ---------------------------------------------------------------------------
extern "C" void kernel_launch(void* const* d_in, const int* in_sizes, int n_in,
                              void* d_out, int out_size, void* d_ws, size_t ws_size,
                              hipStream_t stream)
{
    (void)in_sizes; (void)n_in; (void)out_size; (void)ws_size;
    const int B = GAT_B, N = GAT_N, F = GAT_F;

    const float* slots[2] = { (const float*)d_in[0], (const float*)d_in[1] };
    const float* ne = (const float*)d_in[2];
    const float* W1 = (const float*)d_in[3];
    const float* a1 = (const float*)d_in[4];
    const float* W2 = (const float*)d_in[5];
    const float* a2 = (const float*)d_in[6];

    float* out = (float*)d_out;
    const size_t embSz = (size_t)B * N * F;   // 1,048,576
    const size_t attSz = (size_t)B * N * N;   // 16,777,216
    float* emb[2]    = { out,               out + embSz };
    float* attIn[2]  = { out + 2 * embSz,   out + 2 * embSz + attSz };
    float* attOut[2] = { out + 2 * embSz + 2 * attSz,
                         out + 2 * embSz + 3 * attSz };

    float* ws  = (float*)d_ws;
    float* xw  = ws;                          // B*N*F f32
    float* hb  = xw + embSz;                  // B*N*F f32
    float* s1v = hb + embSz;                  // B*N
    float* s2v = s1v + (size_t)B * N;         // B*N
    __bf16* wt1  = (__bf16*)(s2v + (size_t)B * N);  // F*F bf16 (W1^T)
    __bf16* wt2  = wt1 + (size_t)F * F;             // F*F bf16 (W2^T)
    __bf16* xwT  = wt2 + (size_t)F * F;             // B*F*N bf16 (xw^T)

    // Grids
    const int gemmWaves   = B * (N / 16);     // 512 waves (16x128 tiles)
    const int gemmBlocks  = gemmWaves / 8;    // 64 blocks x 256 thr
    const int scoreBlocks = (B * N) / 8;      // 1024
    const int smRows      = B * N;            // 8192
    const int lsCols      = B * F;            // 512
    const dim3 trW((F * F) / 256, 1);         // W transpose: 64 blocks
    const dim3 trX((N * F) / 256, B);         // xw transpose: 1024 x 4

    // One-time weight transposes (bf16)
    cvt_transpose_bf16<<<trW, 256, 0, stream>>>(W1, wt1, F, F);
    cvt_transpose_bf16<<<trW, 256, 0, stream>>>(W2, wt2, F, F);

    for (int s = 0; s < 2; ++s) {
        // ----- layer 1 -----
        wmma_gemm_bf16T<GAT_F, false><<<gemmBlocks, 256, 0, stream>>>(
            slots[s], wt1, xw, N, F,
            (long)N * F, 0L, (long)N * F);

        scores_kernel<<<scoreBlocks, 256, 0, stream>>>(ne, xw, a1, s1v, s2v);
        attn_softmax_kernel<<<smRows, 256, 0, stream>>>(s1v, s2v, attIn[s]);

        cvt_transpose_bf16<<<trX, 256, 0, stream>>>(xw, xwT, N, F);

        wmma_gemm_bf16T<GAT_N, true><<<gemmBlocks, 256, 0, stream>>>(
            attIn[s], xwT, hb, N, F,
            (long)N * N, (long)F * N, (long)N * F);

        // ----- layer 2 (input already ReLU'd; relu(relu(x)) == relu(x)) -----
        wmma_gemm_bf16T<GAT_F, false><<<gemmBlocks, 256, 0, stream>>>(
            hb, wt2, xw, N, F,
            (long)N * F, 0L, (long)N * F);

        scores_kernel<<<scoreBlocks, 256, 0, stream>>>(ne, xw, a2, s1v, s2v);
        attn_softmax_kernel<<<smRows, 256, 0, stream>>>(s1v, s2v, attOut[s]);

        cvt_transpose_bf16<<<trX, 256, 0, stream>>>(xw, xwT, N, F);

        wmma_gemm_bf16T<GAT_N, true><<<gemmBlocks, 256, 0, stream>>>(
            attOut[s], xwT, hb, N, F,
            (long)N * N, (long)F * N, (long)N * F);

        log_softmax_axis1_kernel<<<lsCols, 256, 0, stream>>>(hb, emb[s]);
    }
}
